// EdgeClassifier_32641751449687
// MI455X (gfx1250) — matrix operand order repaired and checked by reading, not compiled, not run on hardware
//
#include <hip/hip_runtime.h>
#include <hip/hip_bf16.h>

// ---------------------------------------------------------------------------
// EdgeClassifier (hetero GraphSAGE, L=2, H=256) for MI455X / gfx1250.
//  - All H-wide linears: v_wmma_f32_16x16x32_bf16 (bf16 in, f32 accumulate).
//  - Weights pre-packed to bf16 arenas; B tiles staged to LDS with CDNA5
//    global_load_async_to_lds_b128 (ASYNCcnt), double-buffered and software-
//    pipelined so the DMA overlaps the WMMAs of the previous K-chunk.
//  - Aggregation: native global_atomic_add_f32, one channel per lane.
//  - Mean divide fused into GEMM A-staging; edge-MLP concat staged on the fly.
// ---------------------------------------------------------------------------

namespace {
constexpr int kNAcc  = 100000;
constexpr int kNTx   = 300000;
constexpr int kE     = 600000;
constexpr int kH     = 256;
constexpr int kTxIn  = 32;
constexpr int kAccIn = 16;
constexpr int kL     = 2;
constexpr int kM     = 300000;
constexpr float kBnEps = 1e-5f;
constexpr int kEPB   = 16;    // edges per block in aggregation
}

typedef __attribute__((ext_vector_type(16))) __bf16 v16bf;
typedef __attribute__((ext_vector_type(8)))  float  v8f;

__device__ __forceinline__ unsigned short f2bf(float f) {
  unsigned u = __float_as_uint(f);
  u += 0x7FFFu + ((u >> 16) & 1u);       // round-to-nearest-even
  return (unsigned short)(u >> 16);
}

// ---------------------------------------------------------------------------
// Zero fill / inverse-count / weight packing helpers
// ---------------------------------------------------------------------------
__global__ void fill0_kernel(float* __restrict__ p, long n) {
  long i = (long)blockIdx.x * blockDim.x + threadIdx.x;
  long stride = (long)gridDim.x * blockDim.x;
  for (; i < n; i += stride) p[i] = 0.f;
}

__global__ void invcnt_kernel(float* __restrict__ p, int n) {
  int i = blockIdx.x * blockDim.x + threadIdx.x;
  if (i < n) p[i] = 1.f / fmaxf(p[i], 1.f);
}

// Pack [Wl | Wr] (each [256,256] f32, row-major) into bf16 rows of length 512.
__global__ void pack_wcat_kernel(const float* __restrict__ Wl,
                                 const float* __restrict__ Wr,
                                 unsigned short* __restrict__ outp) {
  const int i = blockIdx.x * 256 + threadIdx.x;   // grid 512 -> 131072 elems
  const int h = i >> 9, k = i & 511;
  const float v = (k < kH) ? Wl[h * kH + k] : Wr[h * kH + (k - kH)];
  outp[i] = f2bf(v);
}

__global__ void pack_flat_kernel(const float* __restrict__ W,
                                 unsigned short* __restrict__ outp, int n) {
  const int i = blockIdx.x * 256 + threadIdx.x;
  if (i < n) outp[i] = f2bf(W[i]);
}

// ---------------------------------------------------------------------------
// Input projection: out = relu(X[N,Kin] @ W[256,Kin]^T + b).  Tiny K, VALU FMA.
// ---------------------------------------------------------------------------
__global__ __launch_bounds__(256) void proj_relu_kernel(
    const float* __restrict__ X, const float* __restrict__ W,
    const float* __restrict__ bias, float* __restrict__ out, int Kin) {
  __shared__ float sW[kH * 32];
  __shared__ float sX[32 * 32];
  const int tid = threadIdx.x;
  for (int i = tid; i < kH * Kin; i += 256) sW[i] = W[i];
  const long row0 = (long)blockIdx.x * 32;
  for (int i = tid; i < 32 * Kin; i += 256) sX[i] = X[row0 * Kin + i];
  __syncthreads();
  const float bv = bias[tid];
  for (int r = 0; r < 32; ++r) {
    float s = bv;
    for (int k = 0; k < Kin; ++k) s = fmaf(sX[r * Kin + k], sW[tid * Kin + k], s);
    out[(row0 + r) * kH + tid] = fmaxf(s, 0.f);
  }
}

// ---------------------------------------------------------------------------
// Edge aggregation: agg[dst] += x_src[src]; cnt[dst] += 1.
// ---------------------------------------------------------------------------
__global__ __launch_bounds__(256) void aggregate_kernel(
    const int* __restrict__ src, const int* __restrict__ dst,
    const float* __restrict__ xs, float* __restrict__ agg,
    float* __restrict__ cnt, int E) {
  const int tid = threadIdx.x;
  const int e0 = blockIdx.x * kEPB;
  const int eend = min(e0 + kEPB, E);
  for (int e = e0; e < eend; ++e) {
    const int s = src[e];
    const int d = dst[e];
    if (e + 1 < eend)
      __builtin_prefetch(&xs[(long)src[e + 1] * kH + tid], 0, 1);  // global_prefetch_b8
    unsafeAtomicAdd(&agg[(long)d * kH + tid], xs[(long)s * kH + tid]);
    if (tid == 0) unsafeAtomicAdd(&cnt[d], 1.0f);
  }
}

// ---------------------------------------------------------------------------
// WMMA GEMM: out[N,256] = epilogue( A @ Bbf^T ), Bbf = bf16 [256][KTOT].
//  GMODE==0 (SAGE, KTOT=512): A[n,k] = k<256 ? A1[n,k]*rowscale[n] : A2[n,k-256]
//                             epilogue = BN(bias+v) then ReLU
//  GMODE==1 (edge MLP, KTOT=544): A[n,k] = k<256 ? gfeat[sidx[n],k]
//                                        : k<512 ? gfeat[ridx[n],k-256]
//                                        : graw[n,k-512];  epilogue = bias+ReLU
// Block: 256 thr = 8 waves; tile 32 rows x 256 cols; wave tile 16x64 (4 wmma
// per K-chunk sharing one A fragment).  B is staged with async DMA to LDS,
// double-buffered: chunk i+1's DMA + A staging overlap chunk i's WMMAs.
// ---------------------------------------------------------------------------
template <int KTOT, int GMODE>
__global__ __launch_bounds__(256) void wmma_gemm_kernel(
    const float* __restrict__ A1, const float* __restrict__ rowscale,
    const float* __restrict__ A2,
    const unsigned short* __restrict__ Bbf,
    const float* __restrict__ bias,
    const float* __restrict__ bng, const float* __restrict__ bnb,
    const float* __restrict__ bnm, const float* __restrict__ bnv,
    float* __restrict__ out,
    const int* __restrict__ sidx, const int* __restrict__ ridx,
    const float* __restrict__ gfeat, const float* __restrict__ graw) {
  constexpr int LDA = 36;  // ushorts/row: 72 B  (8B-aligned b64 stores)
  constexpr int LDB = 40;  // ushorts/row: 80 B  (16B-aligned async b128, bank-clean)
  constexpr int NK  = KTOT / 32;
  __shared__ __align__(16) unsigned short lA[2][32 * LDA];
  __shared__ __align__(16) unsigned short lB[2][256 * LDB];
  __shared__ int sRow[2][32];

  const int tid  = threadIdx.x;
  const int lane = tid & 31;
  const int wv   = tid >> 5;
  const int wr   = (wv >> 2) << 4;      // wave row tile: 0 / 16
  const int cg   = (wv & 3) << 6;       // wave col group: 0,64,128,192
  const int half = lane >> 4;
  const int ml   = lane & 15;
  const long row0 = (long)blockIdx.x * 32;

  if (GMODE == 1) {
    if (tid < 32)      sRow[0][tid] = sidx[row0 + tid];
    else if (tid < 64) sRow[1][tid - 32] = ridx[row0 + (tid - 32)];
    __syncthreads();
  }

  const unsigned lBbase0 = (unsigned)(unsigned long long)(size_t)&lB[0][0];
  const unsigned lBbase1 = (unsigned)(unsigned long long)(size_t)&lB[1][0];

  // ---- staging helpers -----------------------------------------------------
  auto stageB = [&](int kk, unsigned base) {
#pragma unroll
    for (int i = 0; i < 4; ++i) {
      const int t = tid + i * 256;          // 0..1023
      const int hrow = t >> 2;              // 0..255
      const int seg = t & 3;                // 16B segment within the 64B row
      const unsigned lds = base + (unsigned)(hrow * (LDB * 2) + seg * 16);
      const unsigned goff = (unsigned)(hrow * (KTOT * 2) + kk * 2 + seg * 16);
      asm volatile("global_load_async_to_lds_b128 %0, %1, %2"
                   :: "v"(lds), "v"(goff), "s"(Bbf) : "memory");
    }
  };
  auto stageA = [&](int kk, unsigned short* dst) {
    const int r = tid >> 3;               // 0..31
    const int k0 = (tid & 7) * 4;         // 0,4,...,28
    const int kg = kk + k0;
    const long n = row0 + r;
    float4 v;
    if (GMODE == 1) {
      if (kg < kH)          v = *(const float4*)&gfeat[(long)sRow[0][r] * kH + kg];
      else if (kg < 2 * kH) v = *(const float4*)&gfeat[(long)sRow[1][r] * kH + (kg - kH)];
      else                  v = *(const float4*)&graw[n * kTxIn + (kg - 2 * kH)];
    } else {
      if (kg < kH) {
        v = *(const float4*)&A1[n * kH + kg];
        const float s = rowscale[n];       // fused mean divide
        v.x *= s; v.y *= s; v.z *= s; v.w *= s;
      } else {
        v = *(const float4*)&A2[n * kH + (kg - kH)];
      }
    }
    union { unsigned short us[4]; unsigned long long u64; } p;
    p.us[0] = f2bf(v.x); p.us[1] = f2bf(v.y);
    p.us[2] = f2bf(v.z); p.us[3] = f2bf(v.w);
    *(unsigned long long*)&dst[r * LDA + k0] = p.u64;
  };

  v8f c[4] = {};

  // ---- fragment math: 4 WMMA from buffer p ---------------------------------
  auto domma = [&](const unsigned short* lAbuf, const unsigned short* lBbuf) {
    union { unsigned u32[8]; v16bf v; } au;
    const unsigned short* arow = &lAbuf[(wr + ml) * LDA];
#pragma unroll
    for (int w = 0; w < 4; ++w) {
      // A 16x32 bf16: lanes 0-15: K 0..7,16..23 ; lanes 16-31: K 8..15,24..31
      au.u32[w]     = *(const unsigned*)&arow[8 * half + 2 * w];
      au.u32[w + 4] = *(const unsigned*)&arow[16 + 8 * half + 2 * w];
    }
#pragma unroll
    for (int t = 0; t < 4; ++t) {
      union { unsigned u32[8]; v16bf v; } bu;
      const unsigned short* brow = &lBbuf[(cg + t * 16 + ml) * LDB];
#pragma unroll
      for (int w = 0; w < 8; ++w) {
        // B 32x16 bf16: lanes 0-15: K 0..7,16..23 ; lanes 16-31: K 8..15,24..31
        bu.u32[w] = *(const unsigned*)&brow[(w < 4 ? 8 * half + 2 * w : 16 + 8 * half + 2 * (w - 4))];
      }
      c[t] = __builtin_amdgcn_wmma_f32_16x16x32_bf16(
          false, au.v, false, bu.v, (short)0, c[t], false, false);
    }
  };

  // ---- pipelined main loop (double-buffered) -------------------------------
  stageB(0, lBbase0);
  stageA(0, &lA[0][0]);
  asm volatile("s_wait_asynccnt 0" ::: "memory");
  __syncthreads();

  for (int i = 0; i < NK; ++i) {
    const int p = i & 1;
    if (i + 1 < NK) {
      // prefetch next chunk into the other buffer: DMA + A staging run
      // concurrently with this chunk's fragment loads and WMMAs.
      stageB((i + 1) * 32, p ? lBbase0 : lBbase1);
      stageA((i + 1) * 32, &lA[p ^ 1][0]);
    }
    domma(&lA[p][0], &lB[p][0]);
    if (i + 1 < NK) {
      asm volatile("s_wait_asynccnt 0" ::: "memory");
      __syncthreads();
    }
  }

  // ---- epilogue: bias (+BN for SAGE), ReLU ----
#pragma unroll
  for (int t = 0; t < 4; ++t) {
    const int col = cg + t * 16 + ml;
    const float bv = bias[col];
    float scale = 1.f, shift = 0.f;
    if (GMODE == 0) {
      const float s = bng[col] * rsqrtf(bnv[col] + kBnEps);
      scale = s;
      shift = bnb[col] - bnm[col] * s;
    }
#pragma unroll
    for (int r = 0; r < 8; ++r) {
      // C/D layout: M = r + 8*half, N = lane&15
      const long row = row0 + wr + r + 8 * half;
      float v = (c[t][r] + bv) * scale + shift;
      v = fmaxf(v, 0.f);
      out[row * kH + col] = v;
    }
  }
}

// ---------------------------------------------------------------------------
// Final: out[m] = dot(h[m,:256], W2) + b2.  One wave32 per row, shfl reduce.
// ---------------------------------------------------------------------------
__global__ __launch_bounds__(256) void final_dot_kernel(
    const float* __restrict__ h, const float* __restrict__ w2,
    const float* __restrict__ b2, float* __restrict__ out, int M) {
  __shared__ float sw[kH];
  const int tid = threadIdx.x;
  sw[tid] = w2[tid];
  __syncthreads();
  const int wv = tid >> 5;
  const int lane = tid & 31;
  const int row = blockIdx.x * 8 + wv;
  if (row >= M) return;
  const float* hr = h + (long)row * kH;
  float s = 0.f;
#pragma unroll
  for (int j = 0; j < 8; ++j) s = fmaf(hr[lane + j * 32], sw[lane + j * 32], s);
#pragma unroll
  for (int off = 16; off > 0; off >>= 1) s += __shfl_xor(s, off, 32);
  if (lane == 0) out[row] = s + b2[0];
}

// ---------------------------------------------------------------------------
// Host-side orchestration
// ---------------------------------------------------------------------------
extern "C" void kernel_launch(void* const* d_in, const int* in_sizes, int n_in,
                              void* d_out, int out_size, void* d_ws, size_t ws_size,
                              hipStream_t stream) {
  (void)in_sizes; (void)n_in; (void)out_size; (void)ws_size;

  const float* x_acc        = (const float*)d_in[0];
  const float* x_tx         = (const float*)d_in[1];
  const float* tx_raw       = (const float*)d_in[2];
  const int*   ei_sends     = (const int*)d_in[3];   // [2,E]: src=+0, dst=+E
  const int*   ei_recv      = (const int*)d_in[4];
  const int*   sender_idx   = (const int*)d_in[5];
  const int*   receiver_idx = (const int*)d_in[6];
  const float* W_acc        = (const float*)d_in[7];
  const float* b_acc        = (const float*)d_in[8];
  const float* W_tx         = (const float*)d_in[9];
  const float* b_tx         = (const float*)d_in[10];
  const float* Wl_at        = (const float*)d_in[11];
  const float* bl_at        = (const float*)d_in[12];
  const float* Wr_at        = (const float*)d_in[13];
  const float* Wl_ta        = (const float*)d_in[14];
  const float* bl_ta        = (const float*)d_in[15];
  const float* Wr_ta        = (const float*)d_in[16];
  const float* bn_g         = (const float*)d_in[17];
  const float* bn_b         = (const float*)d_in[18];
  const float* bn_m         = (const float*)d_in[19];
  const float* bn_v         = (const float*)d_in[20];
  const float* W1           = (const float*)d_in[21];
  const float* b1           = (const float*)d_in[22];
  const float* W2           = (const float*)d_in[23];
  const float* b2           = (const float*)d_in[24];

  // Workspace layout (floats, then bf16 weight arenas)
  const size_t ACC_F = (size_t)kNAcc * kH;   // 25.6M
  const size_t TX_F  = (size_t)kNTx * kH;    // 76.8M
  float* ws      = (float*)d_ws;
  float* acc_a   = ws;
  float* acc_b   = acc_a + ACC_F;
  float* tx_a    = acc_b + ACC_F;
  float* tx_b    = tx_a + TX_F;
  float* agg_acc = tx_b + TX_F;
  float* agg_tx  = agg_acc + ACC_F;
  float* cnt_acc = agg_tx + TX_F;
  float* cnt_tx  = cnt_acc + kNAcc;
  float* h_mid   = agg_tx;                   // reused after conv layers

  unsigned short* wpack = (unsigned short*)(cnt_tx + kNTx);
  const size_t WCAT = (size_t)kH * 512;      // 131072 bf16 per [Wl|Wr] arena
  unsigned short* wcat_at[kL] = { wpack, wpack + WCAT };
  unsigned short* wcat_ta[kL] = { wpack + 2 * WCAT, wpack + 3 * WCAT };
  unsigned short* w1bf = wpack + 4 * WCAT;   // 256*544 bf16

  const long HH = (long)kH * kH;

  // 0) Pack weights to bf16 arenas (feeds the async-DMA B path)
  for (int l = 0; l < kL; ++l) {
    pack_wcat_kernel<<<512, 256, 0, stream>>>(Wl_at + l * HH, Wr_at + l * HH, wcat_at[l]);
    pack_wcat_kernel<<<512, 256, 0, stream>>>(Wl_ta + l * HH, Wr_ta + l * HH, wcat_ta[l]);
  }
  pack_flat_kernel<<<(kH * 544 + 255) / 256, 256, 0, stream>>>(W1, w1bf, kH * 544);

  // 1) Input projections
  proj_relu_kernel<<<kNAcc / 32, 256, 0, stream>>>(x_acc, W_acc, b_acc, acc_a, kAccIn);
  proj_relu_kernel<<<kNTx / 32, 256, 0, stream>>>(x_tx, W_tx, b_tx, tx_a, kTxIn);

  float* acc_cur = acc_a; float* acc_nxt = acc_b;
  float* tx_cur  = tx_a;  float* tx_nxt  = tx_b;

  const int aggBlocks = (kE + kEPB - 1) / kEPB;

  for (int l = 0; l < kL; ++l) {
    // 2a) zero accumulators
    fill0_kernel<<<2048, 256, 0, stream>>>(agg_tx, (long)TX_F);
    fill0_kernel<<<2048, 256, 0, stream>>>(agg_acc, (long)ACC_F);
    fill0_kernel<<<512, 256, 0, stream>>>(cnt_tx, (long)kNTx);
    fill0_kernel<<<512, 256, 0, stream>>>(cnt_acc, (long)kNAcc);

    // 2b) scatter-mean aggregation (atomic f32)
    aggregate_kernel<<<aggBlocks, 256, 0, stream>>>(
        ei_sends, ei_sends + kE, acc_cur, agg_tx, cnt_tx, kE);
    aggregate_kernel<<<aggBlocks, 256, 0, stream>>>(
        ei_recv, ei_recv + kE, tx_cur, agg_acc, cnt_acc, kE);
    invcnt_kernel<<<(kNTx + 255) / 256, 256, 0, stream>>>(cnt_tx, kNTx);
    invcnt_kernel<<<(kNAcc + 255) / 256, 256, 0, stream>>>(cnt_acc, kNAcc);

    // 2c) SAGE linear + BN + ReLU (WMMA, K = 256|256)
    wmma_gemm_kernel<512, 0><<<kNTx / 32, 256, 0, stream>>>(
        agg_tx, cnt_tx, tx_cur, wcat_at[l],
        bl_at + l * kH, bn_g + l * kH, bn_b + l * kH, bn_m + l * kH, bn_v + l * kH,
        tx_nxt, nullptr, nullptr, nullptr, nullptr);
    wmma_gemm_kernel<512, 0><<<kNAcc / 32, 256, 0, stream>>>(
        agg_acc, cnt_acc, acc_cur, wcat_ta[l],
        bl_ta + l * kH, bn_g + l * kH, bn_b + l * kH, bn_m + l * kH, bn_v + l * kH,
        acc_nxt, nullptr, nullptr, nullptr, nullptr);

    { float* t = acc_cur; acc_cur = acc_nxt; acc_nxt = t; }
    { float* t = tx_cur;  tx_cur  = tx_nxt;  tx_nxt  = t; }
  }

  // 3) Edge MLP layer 1: relu([h_s | h_r | tx_raw] @ W1^T + b1), K=544 (WMMA)
  wmma_gemm_kernel<544, 1><<<kM / 32, 256, 0, stream>>>(
      nullptr, nullptr, nullptr, w1bf,
      b1, nullptr, nullptr, nullptr, nullptr,
      h_mid, sender_idx, receiver_idx, acc_cur, tx_raw);

  // 4) Final logit: h @ W2^T + b2
  final_dot_kernel<<<kM / 8, 256, 0, stream>>>(h_mid, W2, b2, (float*)d_out, kM);
}
